// ContinuousAttention_1314259993004
// MI455X (gfx1250) — matrix operand order repaired and checked by reading, not compiled, not run on hardware
//
#include <hip/hip_runtime.h>
#include <hip/hip_bf16.h>
#include <math.h>

#define B_   2
#define NH_  16
#define T_   2048
#define N_   256
#define NHEADS (B_*NH_)          // 32
#define QTILES (T_/16)           // 128

typedef __attribute__((ext_vector_type(16))) __bf16 v16bf;
typedef __attribute__((ext_vector_type(8)))  __bf16 bf16x8;
typedef __attribute__((ext_vector_type(8)))  float  v8f;

// Async global -> LDS copy of 16 bytes per lane (tracked by ASYNCcnt).
#define ASYNC_B128(lds_ptr, gptr)                                            \
  asm volatile("global_load_async_to_lds_b128 %0, %1, off"                   \
               :: "v"((unsigned)(uintptr_t)(lds_ptr)), "v"(gptr) : "memory")

#if __has_builtin(__builtin_amdgcn_s_wait_asynccnt)
#define WAIT_ASYNC(n) __builtin_amdgcn_s_wait_asynccnt(n)
#else
#define WAIT_ASYNC(n) asm volatile("s_wait_asynccnt %0" :: "i"(n) : "memory")
#endif

// ---------------------------------------------------------------------------
// Kernel 1: RoPE(Q) -> bf16, row-major [head][t][n]
// ---------------------------------------------------------------------------
__global__ void rope_bf16_kernel(const float* __restrict__ Q,
                                 __bf16* __restrict__ qr) {
  int idx = blockIdx.x * blockDim.x + threadIdx.x;          // one thread per (even,odd) pair
  if (idx >= NHEADS * T_ * (N_ / 2)) return;
  int np  = idx & (N_ / 2 - 1);                              // pair index 0..127
  int row = idx >> 7;                                        // head*T + t
  int t   = row & (T_ - 1);
  // freqs = theta^(-(2*np)/N) / (2*pi), theta = 2^16
  float freq  = __powf(65536.0f, -(float)(2 * np) / (float)N_) * 0.15915494309189535f;
  float phase = (float)t * freq;
  phase -= floorf(phase);                                    // phases % 1.0
  float ph = phase * 6.283185307179586f;
  float s, c;
  __sincosf(ph, &s, &c);
  const float* qp = Q + (size_t)row * N_ + 2 * np;
  float q0 = qp[0], q1 = qp[1];
  __bf16* o = qr + (size_t)row * N_ + 2 * np;
  o[0] = (__bf16)(q0 * c - q1 * s);                          // even: v*c + (-v_odd)*s
  o[1] = (__bf16)(q1 * c + q0 * s);                          // odd : v*c + ( v_even)*s
}

// ---------------------------------------------------------------------------
// Kernel 2: V (f32, [head][t][n]) -> Vt (bf16, [head][n][t]) via LDS tiles
// ---------------------------------------------------------------------------
__global__ void vt_bf16_kernel(const float* __restrict__ V,
                               __bf16* __restrict__ vt) {
  __shared__ float tile[32][33];
  int h  = blockIdx.z;
  int n0 = blockIdx.x * 32;
  int t0 = blockIdx.y * 32;
  const float* vb = V  + (size_t)h * T_ * N_;
  __bf16*      ob = vt + (size_t)h * T_ * N_;
  int tx = threadIdx.x, ty = threadIdx.y;                    // block (32,8)
  #pragma unroll
  for (int i = 0; i < 32; i += 8)
    tile[ty + i][tx] = vb[(size_t)(t0 + ty + i) * N_ + (n0 + tx)];
  __syncthreads();
  #pragma unroll
  for (int i = 0; i < 32; i += 8)
    ob[(size_t)(n0 + ty + i) * T_ + (t0 + tx)] = (__bf16)tile[tx][ty + i];
}

// ---------------------------------------------------------------------------
// Kernel 3: 4-wave workgroup flash-style two-GEMM attention (no softmax).
//   WG = (head, 4 consecutive query tiles). Key-pair operands (32 QR key rows
//   + 32 Vt columns, 32 KB) are staged in LDS with async global->LDS copies,
//   double-buffered one pair ahead, shared by all 4 waves.
//   GEMM1 computes S^T so its D tile lane-aligns with GEMM2's A fragment.
// ---------------------------------------------------------------------------
__device__ __forceinline__ v8f zero8() {
  v8f z = {0.f, 0.f, 0.f, 0.f, 0.f, 0.f, 0.f, 0.f};
  return z;
}
__device__ __forceinline__ v16bf cat8(bf16x8 lo, bf16x8 hi) {
  return __builtin_shufflevector(lo, hi, 0, 1, 2, 3, 4, 5, 6, 7,
                                 8, 9, 10, 11, 12, 13, 14, 15);
}

__global__ void __launch_bounds__(128, 1)
attn_kernel(const __bf16* __restrict__ qr,
            const __bf16* __restrict__ vt,
            float* __restrict__ out) {
  __shared__ __bf16 sK[2][32 * 256];   // 2 x 16 KB: key rows of the pair
  __shared__ __bf16 sV[2][256 * 32];   // 2 x 16 KB: Vt columns of the pair

  const int head  = blockIdx.x >> 5;              // 32 WGs per head
  const int qbase = (blockIdx.x & 31) * 4;        // 4 query tiles per WG
  const int tid   = threadIdx.x;
  const int wv    = tid >> 5;                     // wave 0..3
  const int lane  = tid & 31;
  const int lm    = lane & 15;
  const int hi    = lane >> 4;
  const int qt    = qbase + wv;

  const __bf16* qrh = qr + (size_t)head * T_ * N_;
  const __bf16* vth = vt + (size_t)head * T_ * N_;

  const int mypairs  = (qt >> 1) + 1;             // this wave's causal pair count
  const int maxpairs = ((qbase + 3) >> 1) + 1;    // workgroup max

  // ---- cooperative async stage of key-pair kp into buffer buf (16 ops/thread)
  auto stage = [&](int kp, int buf) {
    // 32 contiguous QR key rows: linear 16 KB copy
    const __bf16* gK = qrh + (size_t)kp * 32 * N_;
    #pragma unroll
    for (int j = 0; j < 8; ++j) {
      int c = tid + 128 * j;                      // 16-byte chunk id, 0..1023
      ASYNC_B128(&sK[buf][c * 8], gK + c * 8);
    }
    // 256 Vt rows x 32 t-values: 64 B per row
    #pragma unroll
    for (int j = 0; j < 8; ++j) {
      int idx = tid + 128 * j;                    // 0..1023
      int n = idx >> 2, q = idx & 3;
      ASYNC_B128(&sV[buf][n * 32 + q * 8],
                 vth + (size_t)n * T_ + kp * 32 + q * 8);
    }
  };

  // ---- cache query B-fragments (per-wave, from global): lane holds one query
  //      row, two 8-bf16 runs (8*hi, 16+8*hi) per K-chunk of 32.
  const __bf16* qrow = qrh + (size_t)(qt * 16 + lm) * N_;
  v16bf bq[8];
  #pragma unroll
  for (int kc = 0; kc < 8; ++kc) {
    const __bf16* p = qrow + kc * 32 + hi * 8;
    bq[kc] = cat8(*(const bf16x8*)p, *(const bf16x8*)(p + 16));
  }

  // ---- output accumulators: 16 n-tiles x 8 VGPRs (f32)
  v8f o[16];
  #pragma unroll
  for (int nt = 0; nt < 16; ++nt) o[nt] = zero8();

  stage(0, 0);                                    // prefetch first pair

  for (int kp = 0; kp < maxpairs; ++kp) {
    const int buf = kp & 1;
    if (kp + 1 < maxpairs) {                      // prefetch next pair
      stage(kp + 1, buf ^ 1);
      WAIT_ASYNC(16);                             // in-order: pair kp complete
    } else {
      WAIT_ASYNC(0);
    }
    __syncthreads();                              // pair kp visible to all waves

    if (kp < mypairs) {                           // wave-uniform causal skip
      const int ks0 = 2 * kp, ks1 = 2 * kp + 1;
      const bool has1 = (ks1 <= qt);
      v8f st0 = zero8(), st1 = zero8();

      // ---- GEMM1: S^T tiles for key tiles ks0 / ks1 from LDS
      {
        const __bf16* krow0 = &sK[buf][lm * 256 + hi * 8];
        const __bf16* krow1 = &sK[buf][(16 + lm) * 256 + hi * 8];
        #pragma unroll
        for (int kc = 0; kc < 8; ++kc) {
          v16bf A0 = cat8(*(const bf16x8*)(krow0 + kc * 32),
                          *(const bf16x8*)(krow0 + kc * 32 + 16));
          st0 = __builtin_amdgcn_wmma_f32_16x16x32_bf16(
                    false, A0, false, bq[kc], (short)0, st0, false, false);
          if (has1) {
            v16bf A1 = cat8(*(const bf16x8*)(krow1 + kc * 32),
                            *(const bf16x8*)(krow1 + kc * 32 + 16));
            st1 = __builtin_amdgcn_wmma_f32_16x16x32_bf16(
                      false, A1, false, bq[kc], (short)0, st1, false, false);
          }
        }
      }

      // ---- strictly-causal mask on the diagonal tile: S^T element at vgpr r
      //      is (key = r + 8*hi, query = lm); valid iff key < query.
      if (ks0 == qt) {
        #pragma unroll
        for (int r = 0; r < 8; ++r)
          if (r + 8 * hi >= lm) st0[r] = 0.f;
      }
      if (ks1 == qt) {
        #pragma unroll
        for (int r = 0; r < 8; ++r)
          if (r + 8 * hi >= lm) st1[r] = 0.f;
      }

      // ---- build GEMM2 A fragment (16x32) lane-locally:
      //      elem j = S[query=lm][key=8*hi+j] = st0[j]; elem j+8 from st1.
      v16bf a2;
      #pragma unroll
      for (int j = 0; j < 8; ++j) {
        a2[j]     = (__bf16)st0[j];
        a2[j + 8] = (__bf16)st1[j];
      }

      // ---- GEMM2: O += S * V, B fragments from LDS (32 t-values per n-row)
      #pragma unroll
      for (int nt = 0; nt < 16; ++nt) {
        const __bf16* vrow = &sV[buf][(nt * 16 + lm) * 32 + hi * 8];
        v16bf b2 = cat8(*(const bf16x8*)vrow, *(const bf16x8*)(vrow + 16));
        o[nt] = __builtin_amdgcn_wmma_f32_16x16x32_bf16(
                    false, a2, false, b2, (short)0, o[nt], false, false);
      }
    }

    __syncthreads();                              // all waves done reading buf
  }

  // ---- store: D tile element (vgpr r, lane) = row qt*16 + r + 8*hi, col nt*16 + lm
  float* ob = out + (size_t)head * T_ * N_ + (size_t)(qt * 16 + 8 * hi) * N_ + lm;
  #pragma unroll
  for (int nt = 0; nt < 16; ++nt) {
    #pragma unroll
    for (int r = 0; r < 8; ++r)
      ob[(size_t)r * N_ + nt * 16] = o[nt][r];
  }
}

// ---------------------------------------------------------------------------
extern "C" void kernel_launch(void* const* d_in, const int* in_sizes, int n_in,
                              void* d_out, int out_size, void* d_ws, size_t ws_size,
                              hipStream_t stream) {
  const float* Q = (const float*)d_in[0];
  // d_in[1] (K) intentionally unused: reference uses KR = QR
  const float* V = (const float*)d_in[2];
  float* out = (float*)d_out;

  __bf16* qr = (__bf16*)d_ws;                                  // 32 MB
  __bf16* vt = qr + (size_t)NHEADS * T_ * N_;                  // +32 MB

  // 1) RoPE -> bf16
  {
    int pairs = NHEADS * T_ * (N_ / 2);
    rope_bf16_kernel<<<(pairs + 255) / 256, 256, 0, stream>>>(Q, qr);
  }
  // 2) V transpose -> bf16
  {
    dim3 grid(N_ / 32, T_ / 32, NHEADS);
    dim3 block(32, 8, 1);
    vt_bf16_kernel<<<grid, block, 0, stream>>>(V, vt);
  }
  // 3) attention: 4-wave workgroups, one per (head, 4 query tiles)
  {
    dim3 grid(NHEADS * (QTILES / 4), 1, 1);
    attn_kernel<<<grid, 128, 0, stream>>>(qr, vt, out);
  }
}
